// GINE_26594437497281
// MI455X (gfx1250) — compile-verified, matrix operand order checked by reading, not run on hardware
//
#include <hip/hip_runtime.h>
#include <hip/hip_bf16.h>

// GINE forward for MI455X (gfx1250, wave32).
// Edge/node linear layers run on v_wmma_f32_16x16x32_f16 (f16 in, f32 acc).
// Scatter-sum via fp32 L2 atomics; x tables stay resident in 192MB L2.
// All offsets are 32-bit so global loads/atomics use SADDR+voffset addressing.

typedef __attribute__((ext_vector_type(16))) _Float16 v16h;
typedef __attribute__((ext_vector_type(8)))  float    v8f;

#define N_NODES 100000
#define N_EDGES 1600000
#define IN_F    32
#define DIM_F   64
#define BN_EPS  1e-5f

static __device__ __forceinline__ v8f wmma_f16f32(v16h a, v16h b, v8f c) {
  // 8 args: (neg_a, A, neg_b, B, c_mod, C, reuse_a, reuse_b)
  return __builtin_amdgcn_wmma_f32_16x16x32_f16(false, a, false, b, (short)0, c,
                                                false, false);
}

// ---------------------------------------------------------------------------
// Column stats: per-column sum and sum-of-squares (for BatchNorm batch stats).
// Each thread owns one column and a strided slab of rows -> 1 atomic/thread.
// Launch: total threads must be a multiple of F.
// ---------------------------------------------------------------------------
template <int F>
__global__ void col_stats_kernel(const float* __restrict__ x,
                                 float* __restrict__ sums,
                                 float* __restrict__ sumsq) {
  const unsigned t    = blockIdx.x * blockDim.x + threadIdx.x;
  const unsigned T    = gridDim.x * blockDim.x;
  const unsigned c    = t % F;
  const unsigned step = T / F;
  float s = 0.f, q = 0.f;
  for (unsigned r = t / F; r < N_NODES; r += step) {
    float v = x[r * F + c];
    s += v;
    q += v * v;
  }
  atomicAdd(&sums[c], s);
  atomicAdd(&sumsq[c], q);
}

// ---------------------------------------------------------------------------
// BatchNorm apply (training-mode batch stats, biased variance).
// Optionally mirrors the normalized result into the concatenated output.
// ---------------------------------------------------------------------------
template <int F>
__global__ void bn_apply_kernel(const float* __restrict__ x,
                                const float* __restrict__ sums,
                                const float* __restrict__ sumsq,
                                const float* __restrict__ gamma,
                                const float* __restrict__ beta,
                                float* __restrict__ dst,
                                float* __restrict__ dst2, int dst2Stride,
                                int dst2ColOff) {
  const unsigned total  = (unsigned)N_NODES * F;  // <= 6.4M, fits 32-bit
  const unsigned stride = gridDim.x * blockDim.x;
  const float invN      = 1.0f / (float)N_NODES;
  for (unsigned i = blockIdx.x * blockDim.x + threadIdx.x; i < total;
       i += stride) {
    const unsigned c  = i % F;
    const unsigned r  = i / F;
    const float mu = sums[c] * invN;
    const float va = sumsq[c] * invN - mu * mu;
    const float y  = (x[i] - mu) * rsqrtf(va + BN_EPS) * gamma[c] + beta[c];
    dst[i] = y;
    if (dst2) dst2[r * (unsigned)dst2Stride + (unsigned)dst2ColOff + c] = y;
  }
}

// ---------------------------------------------------------------------------
// GINE edge kernel: for a 16-edge tile per wave,
//   msg = relu(x[src] + edge_attr @ eW + eb)   (edge GEMM on WMMA)
//   aggr[dst] += msg                           (fp32 L2 atomics)
// A-operand f16 lane layout (16x32): lanes 0-15 row M=lane, halves K=
// {kb..kb+7, kb+16..kb+23} with kb = 0 (lanes<16) / 8 (lanes>=16).
// B-operand (32xF tile of 16 cols) uses the mirrored layout with N=lane&15.
// C/D layout: lane holds (M = v + (lane>=16 ? 8 : 0), N = lane&15), v=0..7.
// Phase order per tile: issue ALL gathers -> WMMAs -> combine+atomics, so the
// L2 gathers stay pipelined instead of load/wait/atomic lockstep.
// ---------------------------------------------------------------------------
template <int F>
__global__ void gine_edge_kernel(const float* __restrict__ x,          // [N,F]
                                 const float* __restrict__ edge_attr,  // [E,32]
                                 const int* __restrict__ src,
                                 const int* __restrict__ dstI,
                                 const float* __restrict__ eW,         // [32,F]
                                 const float* __restrict__ eb,         // [F]
                                 float* __restrict__ aggr)             // [N,F]
{
  constexpr int NT = F / 16;
  const int lane  = threadIdx.x & 31;
  const int wave  = threadIdx.x >> 5;
  const int nWav  = (blockDim.x >> 5) * gridDim.x;
  const int wId   = (blockDim.x >> 5) * blockIdx.x + wave;
  const unsigned m     = lane & 15;
  const unsigned kb    = (lane < 16) ? 0u : 8u;
  const unsigned dbase = (lane < 16) ? 0u : 8u;

  // Weights + bias: loaded once per wave (tiny, L2-resident), reused over the
  // grid-stride tile loop.
  v16h  bmat[NT];
  float bias[NT];
#pragma unroll
  for (int nt = 0; nt < NT; ++nt) {
    const unsigned n = nt * 16 + m;
#pragma unroll
    for (int j = 0; j < 8; ++j) {
      bmat[nt][j]     = (_Float16)eW[(kb + j) * F + n];
      bmat[nt][8 + j] = (_Float16)eW[(kb + 16 + j) * F + n];
    }
    bias[nt] = eb[n];
  }

  const int numTiles = N_EDGES / 16;  // exact: 1.6M % 16 == 0
  for (int t = wId; t < numTiles; t += nWav) {
    const unsigned e0 = (unsigned)t * 16u;

    // Prefetch next tile's edge_attr row for this lane (speculative).
    __builtin_prefetch(edge_attr + ((unsigned)(t + nWav) * 16u + m) * IN_F, 0, 1);

    // Phase 1a: edge indices for the 8 edge-rows this lane's D covers.
    unsigned si[8], di[8];
#pragma unroll
    for (int v = 0; v < 8; ++v) {
      const unsigned e = e0 + dbase + (unsigned)v;
      si[v] = (unsigned)src[e];
      di[v] = (unsigned)dstI[e];
    }

    // Phase 1b: A operand (16 edges x 32 feats, f32 -> f16).
    const float* row = edge_attr + (e0 + m) * IN_F;
    v16h a;
#pragma unroll
    for (int j = 0; j < 8; ++j) {
      a[j]     = (_Float16)row[kb + j];
      a[8 + j] = (_Float16)row[kb + 16 + j];
    }

    // Phase 1c: issue every x[src] gather for this tile up front (L2-resident;
    // many outstanding loads overlap with the WMMAs below).
    float xg[NT][8];
#pragma unroll
    for (int nt = 0; nt < NT; ++nt) {
      const unsigned col = nt * 16 + m;
#pragma unroll
      for (int v = 0; v < 8; ++v) xg[nt][v] = x[si[v] * F + col];
    }

    // Phase 2: matrix op.
    v8f c[NT];
#pragma unroll
    for (int nt = 0; nt < NT; ++nt) {
      v8f z = {};
      c[nt] = wmma_f16f32(a, bmat[nt], z);
    }

    // Phase 3: bias + gather + relu + scatter.
#pragma unroll
    for (int nt = 0; nt < NT; ++nt) {
      const unsigned col = nt * 16 + m;
#pragma unroll
      for (int v = 0; v < 8; ++v) {
        float msg = c[nt][v] + bias[nt] + xg[nt][v];
        msg = msg > 0.f ? msg : 0.f;
        atomicAdd(&aggr[di[v] * F + col], msg);
      }
    }
  }
}

// ---------------------------------------------------------------------------
// Node GEMM: out = tanh((x [+ aggr]) @ W [+ b]), WMMA-tiled, 16 rows per wave.
// K in {32,64}, FO = 64. N_NODES % 16 == 0, so tiles are exact.
// ---------------------------------------------------------------------------
template <int K, int FO>
__global__ void node_gemm_tanh_kernel(const float* __restrict__ x,     // [N,K]
                                      const float* __restrict__ aggr,  // [N,K]|null
                                      const float* __restrict__ W,     // [K,FO]
                                      const float* __restrict__ b,     // [FO]|null
                                      float* __restrict__ out, int outStride) {
  constexpr int NT = FO / 16;
  constexpr int KT = K / 32;
  const int lane  = threadIdx.x & 31;
  const int wave  = threadIdx.x >> 5;
  const int nWav  = (blockDim.x >> 5) * gridDim.x;
  const int wId   = (blockDim.x >> 5) * blockIdx.x + wave;
  const unsigned m     = lane & 15;
  const unsigned kb    = (lane < 16) ? 0u : 8u;
  const unsigned dbase = (lane < 16) ? 0u : 8u;

  v16h  bmat[KT][NT];
  float bias[NT];
#pragma unroll
  for (int kt = 0; kt < KT; ++kt)
#pragma unroll
    for (int nt = 0; nt < NT; ++nt) {
      const unsigned n = nt * 16 + m;
#pragma unroll
      for (int j = 0; j < 8; ++j) {
        bmat[kt][nt][j]     = (_Float16)W[(kt * 32 + kb + j) * FO + n];
        bmat[kt][nt][8 + j] = (_Float16)W[(kt * 32 + kb + 16 + j) * FO + n];
      }
    }
#pragma unroll
  for (int nt = 0; nt < NT; ++nt) bias[nt] = b ? b[nt * 16 + m] : 0.f;

  const int numTiles = N_NODES / 16;  // exact: 100000 % 16 == 0
  for (int t = wId; t < numTiles; t += nWav) {
    const unsigned r = (unsigned)t * 16u + m;
    v16h a[KT];
#pragma unroll
    for (int kt = 0; kt < KT; ++kt)
#pragma unroll
      for (int j = 0; j < 8; ++j) {
        const unsigned k0 = kt * 32 + kb + (unsigned)j;
        const unsigned k1 = k0 + 16;
        float v0 = x[r * K + k0];
        float v1 = x[r * K + k1];
        if (aggr) {
          v0 += aggr[r * K + k0];
          v1 += aggr[r * K + k1];
        }
        a[kt][j]     = (_Float16)v0;
        a[kt][8 + j] = (_Float16)v1;
      }

    v8f c[NT];
#pragma unroll
    for (int nt = 0; nt < NT; ++nt) c[nt] = (v8f){};
#pragma unroll
    for (int kt = 0; kt < KT; ++kt)
#pragma unroll
      for (int nt = 0; nt < NT; ++nt)
        c[nt] = wmma_f16f32(a[kt], bmat[kt][nt], c[nt]);

#pragma unroll
    for (int nt = 0; nt < NT; ++nt) {
      const unsigned col = nt * 16 + m;
#pragma unroll
      for (int v = 0; v < 8; ++v) {
        const unsigned orow = (unsigned)t * 16u + dbase + (unsigned)v;
        out[orow * (unsigned)outStride + col] = tanhf(c[nt][v] + bias[nt]);
      }
    }
  }
}

// ---------------------------------------------------------------------------
extern "C" void kernel_launch(void* const* d_in, const int* in_sizes, int n_in,
                              void* d_out, int out_size, void* d_ws,
                              size_t ws_size, hipStream_t stream) {
  const float* X        = (const float*)d_in[0];
  const int*   eidx     = (const int*)d_in[1];
  const float* edgeAttr = (const float*)d_in[2];
  const float* bn_in_g  = (const float*)d_in[3];
  const float* bn_in_b  = (const float*)d_in[4];
  const float* eW0      = (const float*)d_in[5];
  const float* eb0      = (const float*)d_in[6];
  const float* W0       = (const float*)d_in[7];
  const float* b0       = (const float*)d_in[8];
  const float* bn0_g    = (const float*)d_in[9];
  const float* bn0_b    = (const float*)d_in[10];
  const float* eW1      = (const float*)d_in[11];
  const float* eb1      = (const float*)d_in[12];
  const float* W1       = (const float*)d_in[13];
  const float* b1       = (const float*)d_in[14];
  const float* bn1_g    = (const float*)d_in[15];
  const float* bn1_b    = (const float*)d_in[16];
  const float* Wfc      = (const float*)d_in[17];

  const int* src = eidx;            // edge_index[0]
  const int* dst = eidx + N_EDGES;  // edge_index[1]

  // Workspace layout (fp32), ~154 MB total.
  float* ws     = (float*)d_ws;
  float* x0     = ws;                                 // N*32
  float* aggr0  = x0 + (size_t)N_NODES * IN_F;        // N*32
  float* h1     = aggr0 + (size_t)N_NODES * IN_F;     // N*64
  float* x1     = h1 + (size_t)N_NODES * DIM_F;       // N*64
  float* aggr1  = x1 + (size_t)N_NODES * DIM_F;       // N*64
  float* h2     = aggr1 + (size_t)N_NODES * DIM_F;    // N*64
  float* x2     = h2 + (size_t)N_NODES * DIM_F;       // N*64
  float* statsA = x2 + (size_t)N_NODES * DIM_F;       // 32+32
  float* statsB = statsA + 64;                        // 64+64
  float* statsC = statsB + 128;                       // 64+64

  float* out = (float*)d_out;  // [N, 192]: x1 | x2 | x3

  // Zero accumulation targets every call (deterministic across graph replays).
  hipMemsetAsync(aggr0, 0, (size_t)N_NODES * IN_F * sizeof(float), stream);
  hipMemsetAsync(aggr1, 0, (size_t)N_NODES * DIM_F * sizeof(float), stream);
  hipMemsetAsync(statsA, 0, (64 + 128 + 128) * sizeof(float), stream);

  // x0 = BN(X)
  col_stats_kernel<IN_F><<<512, 256, 0, stream>>>(X, statsA, statsA + 32);
  bn_apply_kernel<IN_F><<<1024, 256, 0, stream>>>(X, statsA, statsA + 32,
                                                  bn_in_g, bn_in_b, x0,
                                                  nullptr, 0, 0);
  // Layer 0: edge WMMA + gather + relu + scatter, then node GEMM + tanh.
  gine_edge_kernel<IN_F><<<1024, 256, 0, stream>>>(x0, edgeAttr, src, dst,
                                                   eW0, eb0, aggr0);
  node_gemm_tanh_kernel<IN_F, DIM_F><<<512, 256, 0, stream>>>(x0, aggr0, W0,
                                                              b0, h1, DIM_F);
  col_stats_kernel<DIM_F><<<512, 256, 0, stream>>>(h1, statsB, statsB + 64);
  bn_apply_kernel<DIM_F><<<1024, 256, 0, stream>>>(h1, statsB, statsB + 64,
                                                   bn0_g, bn0_b, x1,
                                                   out, 192, 0);
  // Layer 1.
  gine_edge_kernel<DIM_F><<<1024, 256, 0, stream>>>(x1, edgeAttr, src, dst,
                                                    eW1, eb1, aggr1);
  node_gemm_tanh_kernel<DIM_F, DIM_F><<<512, 256, 0, stream>>>(x1, aggr1, W1,
                                                               b1, h2, DIM_F);
  col_stats_kernel<DIM_F><<<512, 256, 0, stream>>>(h2, statsC, statsC + 64);
  bn_apply_kernel<DIM_F><<<1024, 256, 0, stream>>>(h2, statsC, statsC + 64,
                                                   bn1_g, bn1_b, x2,
                                                   out, 192, 64);
  // x3 = tanh(x2 @ Wfc) straight into out[:, 128:192].
  node_gemm_tanh_kernel<DIM_F, DIM_F><<<512, 256, 0, stream>>>(
      x2, nullptr, Wfc, nullptr, out + 128, 192);
}
